// Attention_23459111371049
// MI455X (gfx1250) — matrix operand order repaired and checked by reading, not compiled, not run on hardware
//
#include <hip/hip_runtime.h>
#include <hip/hip_bf16.h>
#include <stdint.h>

typedef __attribute__((ext_vector_type(16))) __bf16 v16bf;
typedef __attribute__((ext_vector_type(8)))  __bf16 v8bf;
typedef __attribute__((ext_vector_type(8)))  float  v8f;
typedef __attribute__((ext_vector_type(4)))  unsigned int v4u;
typedef __attribute__((ext_vector_type(8)))  int v8i;
typedef __attribute__((ext_vector_type(4)))  int v4i;

// Problem constants
static constexpr int  kB     = 16;
static constexpr int  kC     = 768;
static constexpr int  kN     = 4096;          // 64*64 spatial
static constexpr int  kHeads = 12;
static constexpr long kCN    = (long)kC * kN; // per-batch plane (elements)

__device__ __forceinline__ v16bf concat8(v8bf lo, v8bf hi) {
  return __builtin_shufflevector(lo, hi, 0,1,2,3,4,5,6,7,8,9,10,11,12,13,14,15);
}

// ---------------------------------------------------------------------------
// TDM: DMA one 64-row x 32-elem bf16 tile (contiguous dim = 32 elems = 64B)
// from a row-major tensor (row length 768 elems, nRows rows) into LDS.
// D# bitfields per CDNA5 ISA (§8.3/8.4):
//   g0: count=1 | lds_addr | global_addr[56:0] | type=2
//   g1: data_size=1(2B) | tensor_dim0=768 | tensor_dim1=nRows |
//       tile_dim0=32 | tile_dim1=64 | tensor_dim0_stride=768
// ---------------------------------------------------------------------------
__device__ __forceinline__ void tdm_load_tile_64x32_bf16(unsigned ldsOff,
                                                         unsigned long long gaddr,
                                                         int nRows) {
  v4u g0 = { 1u,                                   // count = 1 valid descriptor
             ldsOff,                               // lds_addr
             (unsigned)gaddr,                      // global_addr[31:0]
             (unsigned)((gaddr >> 32) & 0x1FFFFFFull) | 0x80000000u }; // [56:32] | type=2
  v8i g1 = { (int)0x00010000u,                     // data_size = 1 -> 2 bytes
             (int)(768u << 16),                    // tensor_dim0 = 768 (bits 63:48)
             (int)(((unsigned)nRows & 0xFFFFu) << 16), // tensor_dim1 lo16 (bits 95:80)
             (int)(32u << 16),                     // dim1 hi16=0 | tile_dim0 = 32 (127:112)
             64,                                   // tile_dim1 = 64, tile_dim2 = 0
             768,                                  // tensor_dim0_stride lo32
             0, 0 };                               // stride hi, dim1_stride unused (2D)
  v4i z = {0, 0, 0, 0};
#if __has_include(<hip/amd_detail/amd_gfx1250_TDM.h>)
  v8i z8 = {0, 0, 0, 0, 0, 0, 0, 0};
  __builtin_amdgcn_tensor_load_to_lds(g0, g1, z, z, z8, 0);
#else
  __builtin_amdgcn_tensor_load_to_lds(g0, g1, z, z, 0);
#endif
}

// ---------------------------------------------------------------------------
// Transpose + convert: x[b][c=768][n=4096] fp32  ->  xT[b][n=4096][c=768] bf16
// ---------------------------------------------------------------------------
__global__ __launch_bounds__(256)
void transpose_convert_x(const float* __restrict__ x, __bf16* __restrict__ xT) {
  __shared__ float tile[64][65];
  const int n0 = blockIdx.x * 64;
  const int c0 = blockIdx.y * 64;
  const int b  = blockIdx.z;
  const float* xp = x  + (long)b * kCN;
  __bf16*      op = xT + (long)b * kCN;
  const int tn = threadIdx.x & 63;
  const int tg = threadIdx.x >> 6;   // 0..3
  #pragma unroll
  for (int i = 0; i < 16; ++i) {
    const int c = tg * 16 + i;
    tile[c][tn] = xp[(long)(c0 + c) * kN + n0 + tn];
  }
  __syncthreads();
  #pragma unroll
  for (int i = 0; i < 16; ++i) {
    const int n = tg * 16 + i;
    op[(long)(n0 + n) * kC + c0 + tn] = (__bf16)tile[tn][n];
  }
}

__global__ __launch_bounds__(256)
void convert_w(const float* __restrict__ w, __bf16* __restrict__ dst) {
  const int i = blockIdx.x * 256 + threadIdx.x;   // grid sized exactly: 768*768/256
  dst[i] = (__bf16)w[i];
}

// ---------------------------------------------------------------------------
// TDM-staged GEMM: Out[b][m][n] = sum_k Wb[m][k] * XT[b][n][k] + bias[m]
// Wb: [768,768] bf16 row-major (contiguous k).  XT: [4096,768] bf16 (contiguous k).
// 128 threads = 4 waves, 64x64 tile, K-step 32, double-buffered TDM staging.
// OUTK: 0 = bf16 row-major [m][n], 1 = bf16 transposed [n][m], 2 = fp32 [m][n]
// ---------------------------------------------------------------------------
template <int OUTK>
__global__ __launch_bounds__(128)
void gemm_tdm_wmma(const __bf16* __restrict__ Wb,
                   const __bf16* __restrict__ XT, long xBatchStride,
                   const float* __restrict__ bias,
                   void* __restrict__ outPtr, long outBatchStride)
{
  const int n0 = blockIdx.x * 64;
  const int m0 = blockIdx.y * 64;
  const int b  = blockIdx.z;
  const __bf16* Xb = XT + (long)b * xBatchStride;

  __shared__ __align__(16) __bf16 lsA[2][64 * 32];  // [m][k]
  __shared__ __align__(16) __bf16 lsB[2][64 * 32];  // [n][k]

  const int tid  = threadIdx.x;
  const int lane = tid & 31;
  const int wv   = tid >> 5;
  const int lrow = lane & 15;
  const int lhi  = lane >> 4;

  const unsigned aOff[2] = { (unsigned)(uintptr_t)&lsA[0][0],
                             (unsigned)(uintptr_t)&lsA[1][0] };
  const unsigned bOff[2] = { (unsigned)(uintptr_t)&lsB[0][0],
                             (unsigned)(uintptr_t)&lsB[1][0] };
  const unsigned long long gA = (unsigned long long)(uintptr_t)(Wb + (long)m0 * kC);
  const unsigned long long gB = (unsigned long long)(uintptr_t)(Xb + (long)n0 * kC);

  constexpr int KIT = kC / 32;   // 24
  if (wv == 0) {                 // one TDM issue per instruction (EXEC ignored)
    tdm_load_tile_64x32_bf16(aOff[0], gA, kC);
    tdm_load_tile_64x32_bf16(bOff[0], gB, kN);
  }

  v8f acc[4] = {};
  for (int it = 0; it < KIT; ++it) {
    if (wv == 0) __builtin_amdgcn_s_wait_tensorcnt(0);
    __syncthreads();
    const int cur = it & 1;
    if (wv == 0 && (it + 1) < KIT) {   // prefetch next K-tile into other buffer
      const unsigned long long koff = (unsigned long long)(it + 1) * 64; // 32 elems * 2B
      tdm_load_tile_64x32_bf16(aOff[cur ^ 1], gA + koff, kC);
      tdm_load_tile_64x32_bf16(bOff[cur ^ 1], gB + koff, kN);
    }
    const __bf16* As = &lsA[cur][0];
    const __bf16* Bs = &lsB[cur][0];
    const int arow = wv * 16 + lrow;
    const v16bf afrag = concat8(*reinterpret_cast<const v8bf*>(As + arow * 32 + lhi * 8),
                                *reinterpret_cast<const v8bf*>(As + arow * 32 + 16 + lhi * 8));
    #pragma unroll
    for (int j = 0; j < 4; ++j) {
      const __bf16* bp = Bs + (j * 16 + lrow) * 32 + lhi * 8;
      const v16bf bfrag = concat8(*reinterpret_cast<const v8bf*>(bp),
                                  *reinterpret_cast<const v8bf*>(bp + 16));
      acc[j] = __builtin_amdgcn_wmma_f32_16x16x32_bf16(
          false, afrag, false, bfrag, (short)0, acc[j], false, false);
    }
  }

  // ---- epilogue
  if constexpr (OUTK == 0) {
    __bf16* Out = (__bf16*)outPtr + (long)b * outBatchStride;
    #pragma unroll
    for (int j = 0; j < 4; ++j) {
      const int n = n0 + j * 16 + lrow;
      #pragma unroll
      for (int i = 0; i < 8; ++i) {
        const int m = m0 + wv * 16 + lhi * 8 + i;
        Out[(long)m * kN + n] = (__bf16)(acc[j][i] + bias[m]);
      }
    }
  } else if constexpr (OUTK == 1) {    // transposed: Out[n][m], row stride kC
    __bf16* Out = (__bf16*)outPtr + (long)b * outBatchStride;
    #pragma unroll
    for (int j = 0; j < 4; ++j) {
      const int n = n0 + j * 16 + lrow;
      v8bf ov;
      #pragma unroll
      for (int i = 0; i < 8; ++i)
        ov[i] = (__bf16)(acc[j][i] + bias[m0 + wv * 16 + lhi * 8 + i]);
      *reinterpret_cast<v8bf*>(&Out[(long)n * kC + m0 + wv * 16 + lhi * 8]) = ov;
    }
  } else {
    float* Out = (float*)outPtr + (long)b * outBatchStride;
    #pragma unroll
    for (int j = 0; j < 4; ++j) {
      const int n = n0 + j * 16 + lrow;
      #pragma unroll
      for (int i = 0; i < 8; ++i) {
        const int m = m0 + wv * 16 + lhi * 8 + i;
        Out[(long)m * kN + n] = acc[j][i] + bias[m];
      }
    }
  }
}

// ---------------------------------------------------------------------------
// Fused channel attention per (batch, head).
// Q,K bf16 [64 ch][4096 px] row-major; V transposed vT[b][n=4096][c=768].
// E = Q K^T (contraction over pixels) -> softmax rows -> out = A V,
// output stored transposed into aoutT[b][n][c] for the final TDM GEMM.
// ---------------------------------------------------------------------------
__global__ __launch_bounds__(128)
void attn_channel_wmma(const __bf16* __restrict__ qk,    // [B][2][768][4096]
                       const __bf16* __restrict__ vT,    // [B][4096][768]
                       __bf16*       __restrict__ aoutT) // [B][4096][768]
{
  const int b = blockIdx.x / kHeads;
  const int h = blockIdx.x % kHeads;
  const __bf16* Q  = qk + (long)b * 2 * kCN + (long)(h * 64) * kN;
  const __bf16* K  = Q + kCN;
  const __bf16* Vt = vT    + (long)b * kCN;
  __bf16*       Ot = aoutT + (long)b * kCN;

  __shared__ __align__(16) float  lsE   [64 * 64];  // energy (f32)
  __shared__ __align__(16) __bf16 lsAttn[64 * 64];  // softmaxed attn (bf16)

  const int tid  = threadIdx.x;
  const int lane = tid & 31;
  const int wv   = tid >> 5;
  const int lrow = lane & 15;
  const int lhi  = lane >> 4;

  // ---- Stage 1: E = Q K^T over n = 0..4095, step 32
  v8f eacc[4] = {};
  const int qrow = wv * 16 + lrow;
  for (int n0 = 0; n0 < kN; n0 += 32) {
    const __bf16* qp = &Q[(long)qrow * kN + n0];
    const v16bf afrag = concat8(*reinterpret_cast<const v8bf*>(qp + lhi * 8),
                                *reinterpret_cast<const v8bf*>(qp + 16 + lhi * 8));
    #pragma unroll
    for (int j = 0; j < 4; ++j) {
      const __bf16* kp = &K[(long)(j * 16 + lrow) * kN + n0 + lhi * 8];
      const v16bf bfrag = concat8(*reinterpret_cast<const v8bf*>(kp),
                                  *reinterpret_cast<const v8bf*>(kp + 16));
      eacc[j] = __builtin_amdgcn_wmma_f32_16x16x32_bf16(
          false, afrag, false, bfrag, (short)0, eacc[j], false, false);
    }
  }
  #pragma unroll
  for (int j = 0; j < 4; ++j)
    #pragma unroll
    for (int i = 0; i < 8; ++i)
      lsE[(wv * 16 + lhi * 8 + i) * 64 + j * 16 + lrow] = eacc[j][i] * 0.125f; // HEAD_DIM^-0.5
  __syncthreads();

  // ---- softmax per row (threads 0..63 own one row each)
  if (tid < 64) {
    float mx = -3.4e38f;
    for (int j = 0; j < 64; ++j) mx = fmaxf(mx, lsE[tid * 64 + j]);
    float s = 0.f;
    for (int j = 0; j < 64; ++j) s += __expf(lsE[tid * 64 + j] - mx);
    const float inv = 1.f / s;
    for (int j = 0; j < 64; ++j)
      lsAttn[tid * 64 + j] = (__bf16)(__expf(lsE[tid * 64 + j] - mx) * inv);
  }
  __syncthreads();

  // ---- Stage 2: out = Attn(64x64) @ V(64x4096); V-frags straight from vT
  v16bf af[2];
  #pragma unroll
  for (int kb = 0; kb < 2; ++kb) {
    const __bf16* ap = &lsAttn[(wv * 16 + lrow) * 64 + kb * 32];
    af[kb] = concat8(*reinterpret_cast<const v8bf*>(ap + lhi * 8),
                     *reinterpret_cast<const v8bf*>(ap + 16 + lhi * 8));
  }

  for (int n0 = 0; n0 < kN; n0 += 64) {
    v8f oacc[4] = {};
    #pragma unroll
    for (int j = 0; j < 4; ++j) {
      const int n = n0 + j * 16 + lrow;
      const __bf16* vp = &Vt[(long)n * kC + h * 64];
      #pragma unroll
      for (int kb = 0; kb < 2; ++kb) {
        const __bf16* bp = vp + kb * 32 + lhi * 8;
        const v16bf bfrag = concat8(*reinterpret_cast<const v8bf*>(bp),
                                    *reinterpret_cast<const v8bf*>(bp + 16));
        oacc[j] = __builtin_amdgcn_wmma_f32_16x16x32_bf16(
            false, af[kb], false, bfrag, (short)0, oacc[j], false, false);
      }
    }
    // transposed store: aoutT[n][h*64 + c], 8 consecutive channels per lane
    #pragma unroll
    for (int j = 0; j < 4; ++j) {
      const int n = n0 + j * 16 + lrow;
      v8bf ov;
      #pragma unroll
      for (int i = 0; i < 8; ++i) ov[i] = (__bf16)oacc[j][i];
      *reinterpret_cast<v8bf*>(&Ot[(long)n * kC + h * 64 + wv * 16 + lhi * 8]) = ov;
    }
  }
}

// ---------------------------------------------------------------------------
extern "C" void kernel_launch(void* const* d_in, const int* in_sizes, int n_in,
                              void* d_out, int out_size, void* d_ws, size_t ws_size,
                              hipStream_t stream) {
  (void)in_sizes; (void)n_in; (void)out_size; (void)ws_size;
  const float* x  = (const float*)d_in[0];
  const float* wq = (const float*)d_in[1];
  const float* bq = (const float*)d_in[2];
  const float* wk = (const float*)d_in[3];
  const float* bk = (const float*)d_in[4];
  const float* wvp= (const float*)d_in[5];
  const float* bv = (const float*)d_in[6];
  const float* wo = (const float*)d_in[7];
  const float* bo = (const float*)d_in[8];
  float* out = (float*)d_out;

  // Workspace layout (bf16 elements):
  //  [0]                xT / aoutT alias : kB*kCN           (96 MiB)
  //  [kB*kCN]           weights bf16     : 4 * 768*768      (4.5 MiB)
  //  [.. + 4*KK]        q,k planes       : kB * 2 * kCN     (192 MiB)
  //  [..]               vT               : kB * kCN         (96 MiB)
  const long kKK = (long)kC * kC;
  __bf16* xT   = (__bf16*)d_ws;                 // aliased with aoutT
  __bf16* wbf  = xT  + (long)kB * kCN;          // wq,wk,wv,wo bf16
  __bf16* qk   = wbf + 4 * kKK;
  __bf16* vT   = qk  + (long)kB * 2 * kCN;
  __bf16* aoutT= xT;

  dim3 blk128(128);
  dim3 gGemm(kN / 64, kC / 64, kB);

  // 1) precision/layout prep
  convert_w<<<dim3((unsigned)(kKK / 256)), dim3(256), 0, stream>>>(wq,  wbf + 0 * kKK);
  convert_w<<<dim3((unsigned)(kKK / 256)), dim3(256), 0, stream>>>(wk,  wbf + 1 * kKK);
  convert_w<<<dim3((unsigned)(kKK / 256)), dim3(256), 0, stream>>>(wvp, wbf + 2 * kKK);
  convert_w<<<dim3((unsigned)(kKK / 256)), dim3(256), 0, stream>>>(wo,  wbf + 3 * kKK);
  transpose_convert_x<<<dim3(kN / 64, kC / 64, kB), dim3(256), 0, stream>>>(x, xT);

  // 2) Q, K (row-major out) and V (transposed out) projections via TDM GEMM
  gemm_tdm_wmma<0><<<gGemm, blk128, 0, stream>>>(wbf + 0 * kKK, xT, kCN, bq, qk,       2 * kCN);
  gemm_tdm_wmma<0><<<gGemm, blk128, 0, stream>>>(wbf + 1 * kKK, xT, kCN, bk, qk + kCN, 2 * kCN);
  gemm_tdm_wmma<1><<<gGemm, blk128, 0, stream>>>(wbf + 2 * kKK, xT, kCN, bv, vT,       kCN);

  // 3) fused channel attention (writes transposed aoutT, aliasing xT)
  attn_channel_wmma<<<dim3(kB * kHeads), blk128, 0, stream>>>(qk, vT, aoutT);

  // 4) output projection -> fp32 d_out
  gemm_tdm_wmma<2><<<gGemm, blk128, 0, stream>>>(wbf + 3 * kKK, aoutT, kCN, bo, out, kCN);
}
